// SwinTransformerBlock3D_39170101739821
// MI455X (gfx1250) — matrix operand order repaired and checked by reading, not compiled, not run on hardware
//
#include <hip/hip_runtime.h>

// ---------------------------------------------------------------------------
// Swin Transformer 3D block on gfx1250 (wave32, WMMA bf16 -> f32 accumulate)
// B=1 D=16 H=64 W=64 C=128, ws=8 (N=512 tok/window, 128 windows), NH=4 hd=32
// All fragments are assembled from contiguous 16B chunks (2x uint4 per frag).
// ---------------------------------------------------------------------------

typedef __attribute__((ext_vector_type(16))) __bf16 v16bf;
typedef __attribute__((ext_vector_type(8)))  float  v8f;

#define NWIN 128
#define NTOK 512
#define CDIM 128
#define NHEAD 4
#define HD 32
#define TOKS 65536   // total tokens B*D*H*W
#define APAD 40      // padded LDS row stride (bf16) -> 80B, breaks bank conflicts

static __device__ __forceinline__ v8f wmma_bf16(v16bf a, v16bf b, v8f c) {
  return __builtin_amdgcn_wmma_f32_16x16x32_bf16(false, a, false, b, (short)0, c,
                                                 false, false);
}

// Build a 16-element bf16 fragment from two contiguous 16-byte chunks.
// A-frag (16x32): elem i -> K = i + 16*half   (contiguous run of 16)
//   => chunks at (base + 16*half) and (base + 16*half + 8)
// B-frag (32x16): elem i -> K = i + 16*half   (contiguous run of 16)
//   => chunks at (base + 16*half) and (base + 16*half + 8)
static __device__ __forceinline__ v16bf frag16(const __bf16* p0, const __bf16* p1) {
  union { uint4 u[2]; v16bf v; } f;
  f.u[0] = *(const uint4*)p0;
  f.u[1] = *(const uint4*)p1;
  return f.v;
}

// ----- shared 128x128-output-tile GEMM: C[128,128] = A[128,K] * W^T ---------
// W is the natural [N][K] row-major weight (i.e. column-major B).
// 256 threads = 8 waves in a 4(M) x 2(N) grid; each wave owns 2x4 16x16 frags.
struct Acc { v8f f[2][4]; };

static __device__ __forceinline__ void gemm128(const __bf16* __restrict__ Ag, int lda,
                                               const __bf16* __restrict__ Wg, int ldk,
                                               int ncol0, int K, Acc& acc) {
  __shared__ __align__(16) __bf16 As[128 * APAD];  // [row][k] (32 used of APAD)
  __shared__ __align__(16) __bf16 Bs[128 * APAD];  // [col][k] (transposed tile)
  const int tid  = threadIdx.x;
  const int wave = tid >> 5;
  const int wr   = wave & 3;   // M direction, 32 rows each
  const int wc   = wave >> 2;  // N direction, 64 cols each
  const int lane = tid & 31, half = lane >> 4, nloc = lane & 15;
  const v8f z = {0.f, 0.f, 0.f, 0.f, 0.f, 0.f, 0.f, 0.f};
#pragma unroll
  for (int mi = 0; mi < 2; ++mi)
#pragma unroll
    for (int ni = 0; ni < 4; ++ni) acc.f[mi][ni] = z;

  for (int k0 = 0; k0 < K; k0 += 32) {
    __syncthreads();
    {  // stage A tile 128x32: each thread one 32B chunk
      int base = tid * 16;
      int r = base >> 5, c = base & 31;
      const uint4* src = (const uint4*)(Ag + (size_t)r * lda + k0 + c);
      uint4* dst = (uint4*)(As + r * APAD + c);
      dst[0] = src[0];
      dst[1] = src[1];
    }
    {  // stage B tile as [col][k]: W rows are already K-contiguous
      int colL = tid >> 1, kk = (tid & 1) * 16;
      const uint4* src = (const uint4*)(Wg + (size_t)(ncol0 + colL) * ldk + k0 + kk);
      uint4* dst = (uint4*)(Bs + colL * APAD + kk);
      dst[0] = src[0];
      dst[1] = src[1];
    }
    __syncthreads();

    v16bf bfr[4];
#pragma unroll
    for (int ni = 0; ni < 4; ++ni) {
      const __bf16* bp = Bs + (wc * 64 + ni * 16 + nloc) * APAD + 16 * half;
      bfr[ni] = frag16(bp, bp + 8);
    }
#pragma unroll
    for (int mi = 0; mi < 2; ++mi) {
      const __bf16* ap = As + (wr * 32 + mi * 16 + nloc) * APAD + 16 * half;
      v16bf a = frag16(ap, ap + 8);
#pragma unroll
      for (int ni = 0; ni < 4; ++ni)
        acc.f[mi][ni] = wmma_bf16(a, bfr[ni], acc.f[mi][ni]);
    }
  }
  __syncthreads();
}

// ----- K0: weights -> bf16 (natural [N][K] layout) + RPE table --------------
#define W_TOTAL (49152 + 16384 + 65536 + 65536)
#define PREP_TOTAL (W_TOTAL + 1048576)
__global__ __launch_bounds__(256) void k_prep(
    const float* __restrict__ qkv_w, const float* __restrict__ proj_w,
    const float* __restrict__ m1, const float* __restrict__ m2,
    const int* __restrict__ rel_index, const float* __restrict__ rel_bias,
    __bf16* Wq, __bf16* Wp, __bf16* W1, __bf16* W2, float* rpe) {
  int i = blockIdx.x * 256 + threadIdx.x;
  if (i >= PREP_TOTAL) return;
  if (i < 49152) {
    Wq[i] = (__bf16)qkv_w[i];                       // [384][128]
  } else if (i < 49152 + 16384) {
    int j = i - 49152;
    Wp[j] = (__bf16)proj_w[j];                      // [128][128]
  } else if (i < 49152 + 16384 + 65536) {
    int j = i - (49152 + 16384);
    W1[j] = (__bf16)m1[j];                          // [512][128]
  } else if (i < W_TOTAL) {
    int j = i - (49152 + 16384 + 65536);
    W2[j] = (__bf16)m2[j];                          // [128][512]
  } else {                                          // rpe[h][n][m]
    int j = i - W_TOTAL;
    int idx = j & 262143;
    rpe[j] = rel_bias[rel_index[idx] * 4 + (j >> 18)];
  }
}

// ----- K1: LN1 + roll(-4) + window partition -> xw bf16 [128,512,128] -------
__global__ __launch_bounds__(256) void k_ln1(const float* __restrict__ x,
                                             const float* __restrict__ g,
                                             const float* __restrict__ b,
                                             __bf16* __restrict__ xw) {
  int gw   = blockIdx.x * 8 + (threadIdx.x >> 5);  // one wave per token
  int lane = threadIdx.x & 31;
  int w = gw >> 9, tok = gw & 511;
  int wd = w >> 6, wh = (w >> 3) & 7, ww = w & 7;
  int id = tok >> 6, ih = (tok >> 3) & 7, iw = tok & 7;
  int d  = ((wd * 8 + id) + 4) & 15;   // roll by -ss: read x[(i+ss)%D]
  int h  = ((wh * 8 + ih) + 4) & 63;
  int wx = ((ww * 8 + iw) + 4) & 63;
  const float* src = x + (((size_t)d * 64 + h) * 64 + wx) * CDIM;
  float4 v = ((const float4*)src)[lane];
  float s  = v.x + v.y + v.z + v.w;
  float ss = v.x * v.x + v.y * v.y + v.z * v.z + v.w * v.w;
#pragma unroll
  for (int m = 16; m; m >>= 1) {
    s  += __shfl_xor(s, m, 32);
    ss += __shfl_xor(ss, m, 32);
  }
  float mean = s * (1.f / 128.f);
  float var  = ss * (1.f / 128.f) - mean * mean;
  float inv  = rsqrtf(var + 1e-5f);
  int c = lane * 4;
  union { __bf16 h4[4]; uint2 u; } pk;
  pk.h4[0] = (__bf16)((v.x - mean) * inv * g[c + 0] + b[c + 0]);
  pk.h4[1] = (__bf16)((v.y - mean) * inv * g[c + 1] + b[c + 1]);
  pk.h4[2] = (__bf16)((v.z - mean) * inv * g[c + 2] + b[c + 2]);
  pk.h4[3] = (__bf16)((v.w - mean) * inv * g[c + 3] + b[c + 3]);
  *(uint2*)(xw + (size_t)gw * CDIM + c) = pk.u;
}

// ----- K2: QKV GEMM + head split (q pre-scaled, V stored transposed) --------
__global__ __launch_bounds__(256) void k_qkv(const __bf16* __restrict__ xw,
                                             const __bf16* __restrict__ Wq,
                                             const float* __restrict__ qkv_b,
                                             __bf16* qb, __bf16* kb, __bf16* vt) {
  int mt = blockIdx.x;  // row tile of 128 within the 512-token window
  int nt = blockIdx.y;  // col tile of 128 within 384
  int w  = blockIdx.z;
  const __bf16* Ag = xw + (size_t)w * NTOK * CDIM + (size_t)mt * 128 * CDIM;
  Acc acc;
  gemm128(Ag, CDIM, Wq, CDIM, nt * 128, CDIM, acc);

  int tid = threadIdx.x, wave = tid >> 5, wr = wave & 3, wc = wave >> 2;
  int lane = tid & 31, half = lane >> 4, nloc = lane & 15;
  const float scale = 0.17677669529663687f;  // 1/sqrt(32)
#pragma unroll
  for (int mi = 0; mi < 2; ++mi)
#pragma unroll
    for (int ni = 0; ni < 4; ++ni) {
      int ncol  = nt * 128 + wc * 64 + ni * 16 + nloc;  // 0..383
      int which = ncol >> 7;
      int head  = (ncol >> 5) & 3;
      int di    = ncol & 31;
      float bias = qkv_b[ncol];
      int tok0 = mt * 128 + wr * 32 + mi * 16 + 8 * half;
      if (which == 2) {  // V transposed: [w][head][hd][tok], 8 contiguous toks
        union { __bf16 h8[8]; uint4 u; } pk;
#pragma unroll
        for (int r = 0; r < 8; ++r) pk.h8[r] = (__bf16)(acc.f[mi][ni][r] + bias);
        *(uint4*)(vt + (((size_t)w * NHEAD + head) * HD + di) * NTOK + tok0) = pk.u;
      } else {
        __bf16* dst = (which == 0) ? qb : kb;
#pragma unroll
        for (int r = 0; r < 8; ++r) {
          float vv = acc.f[mi][ni][r] + bias;
          if (which == 0) vv *= scale;
          dst[(((size_t)w * NHEAD + head) * NTOK + tok0 + r) * HD + di] = (__bf16)vv;
        }
      }
    }
}

// ----- K4: attention per (window, head); 16-row strips, exact softmax -------
#define PPAD 520  // padded P row stride (bf16): 1040B, 16B aligned, conflict-free
__global__ __launch_bounds__(256) void k_attn(const __bf16* __restrict__ qb,
                                              const __bf16* __restrict__ kb,
                                              const __bf16* __restrict__ vt,
                                              const float* __restrict__ rpe,
                                              const float* __restrict__ mask,
                                              __bf16* __restrict__ ao) {
  __shared__ __align__(16) float  Ssm[16 * 512];    // scores / PV partials
  __shared__ __align__(16) __bf16 Psm[16 * PPAD];   // softmax probs (A operand)
  __shared__ __align__(16) __bf16 Qsm[16 * APAD];
  int w = blockIdx.x >> 2, head = blockIdx.x & 3;
  const __bf16* Q  = qb + ((size_t)w * NHEAD + head) * NTOK * HD;
  const __bf16* Kp = kb + ((size_t)w * NHEAD + head) * NTOK * HD;
  const __bf16* Vt = vt + ((size_t)w * NHEAD + head) * HD * NTOK;  // [hd][tok]
  const float* rp = rpe + (size_t)head * NTOK * NTOK;
  const float* mk = mask + (size_t)w * NTOK * NTOK;
  int tid = threadIdx.x, lane = tid & 31, wave = tid >> 5;
  int half = lane >> 4, nloc = lane & 15;

  for (int n0 = 0; n0 < NTOK; n0 += 16) {
    __syncthreads();
    if (tid < 64) {  // stage Q strip [16][32] -> padded LDS, 16B chunks
      int r = tid >> 2, q4 = tid & 3;
      ((uint4*)(Qsm + r * APAD))[q4] = ((const uint4*)(Q + (size_t)(n0 + r) * HD))[q4];
    }
    __syncthreads();

    // S strip [16,512] = Q[16,32] * K^T ; each wave does 4 of 32 col-tiles
    {
      const __bf16* ap = Qsm + nloc * APAD + 16 * half;
      v16bf a = frag16(ap, ap + 8);
#pragma unroll
      for (int jj = 0; jj < 4; ++jj) {
        int jt = wave * 4 + jj;
        const __bf16* kpp = Kp + (size_t)(jt * 16 + nloc) * HD + 16 * half;
        v16bf bfr = frag16(kpp, kpp + 8);
        v8f c = {0.f, 0.f, 0.f, 0.f, 0.f, 0.f, 0.f, 0.f};
        c = wmma_bf16(a, bfr, c);
#pragma unroll
        for (int r = 0; r < 8; ++r)
          Ssm[(r + 8 * half) * 512 + jt * 16 + nloc] = c[r];
      }
    }
    __syncthreads();

    // + rpe + mask, then exact row softmax (16 threads per row, 32 cols each)
    {
      int row = tid >> 4, sub = tid & 15;
      const float* rprow = rp + (size_t)(n0 + row) * NTOK;
      const float* mkrow = mk + (size_t)(n0 + row) * NTOK;
      float mx = -1e30f;
      for (int c0 = sub * 32; c0 < sub * 32 + 32; ++c0) {
        float sv = Ssm[row * 512 + c0] + rprow[c0] + mkrow[c0];
        Ssm[row * 512 + c0] = sv;
        mx = fmaxf(mx, sv);
      }
#pragma unroll
      for (int m = 8; m; m >>= 1) mx = fmaxf(mx, __shfl_xor(mx, m, 16));
      float sum = 0.f;
      for (int c0 = sub * 32; c0 < sub * 32 + 32; ++c0) {
        float e = __expf(Ssm[row * 512 + c0] - mx);
        Ssm[row * 512 + c0] = e;
        sum += e;
      }
#pragma unroll
      for (int m = 8; m; m >>= 1) sum += __shfl_xor(sum, m, 16);
      float inv = 1.f / sum;
      for (int c0 = sub * 32; c0 < sub * 32 + 32; ++c0)
        Psm[row * PPAD + c0] = (__bf16)(Ssm[row * 512 + c0] * inv);
    }
    __syncthreads();

    // O[16,32] = P[16,512] * V[512,32]; waves split K (64 each), LDS-reduce
    {
      v8f oacc[2];
      oacc[0] = (v8f){0.f, 0.f, 0.f, 0.f, 0.f, 0.f, 0.f, 0.f};
      oacc[1] = oacc[0];
#pragma unroll
      for (int ks = 0; ks < 2; ++ks) {
        int kbase = wave * 64 + ks * 32;
        const __bf16* ap = Psm + nloc * PPAD + kbase + 16 * half;
        v16bf a = frag16(ap, ap + 8);
#pragma unroll
        for (int ct = 0; ct < 2; ++ct) {
          const __bf16* vp = Vt + (size_t)(ct * 16 + nloc) * NTOK + kbase + 16 * half;
          v16bf bfr = frag16(vp, vp + 8);
          oacc[ct] = wmma_bf16(a, bfr, oacc[ct]);
        }
      }
#pragma unroll
      for (int ct = 0; ct < 2; ++ct)
#pragma unroll
        for (int r = 0; r < 8; ++r)
          Ssm[wave * 512 + (r + 8 * half) * 32 + ct * 16 + nloc] = oacc[ct][r];
    }
    __syncthreads();
    {  // reduce 8 partials; pack 2 bf16 per store
      int row = tid >> 4, c = (tid & 15) * 2;
      float s0 = 0.f, s1 = 0.f;
#pragma unroll
      for (int pw = 0; pw < 8; ++pw) {
        s0 += Ssm[pw * 512 + row * 32 + c];
        s1 += Ssm[pw * 512 + row * 32 + c + 1];
      }
      union { __bf16 h2[2]; unsigned u; } pk;
      pk.h2[0] = (__bf16)s0;
      pk.h2[1] = (__bf16)s1;
      *(unsigned*)(ao + ((size_t)w * NTOK + n0 + row) * CDIM + head * HD + c) = pk.u;
    }
  }
}

// ----- K5: proj GEMM + window reverse + roll(+4) + residual -> d_out --------
__global__ __launch_bounds__(256) void k_proj(const __bf16* __restrict__ ao,
                                              const __bf16* __restrict__ Wp,
                                              const float* __restrict__ proj_b,
                                              const float* __restrict__ x,
                                              float* __restrict__ out) {
  int mt = blockIdx.x, w = blockIdx.z;
  const __bf16* Ag = ao + (size_t)w * NTOK * CDIM + (size_t)mt * 128 * CDIM;
  Acc acc;
  gemm128(Ag, CDIM, Wp, CDIM, 0, CDIM, acc);

  int tid = threadIdx.x, wave = tid >> 5, wr = wave & 3, wc = wave >> 2;
  int lane = tid & 31, half = lane >> 4, nloc = lane & 15;
  int wd = w >> 6, wh = (w >> 3) & 7, ww = w & 7;
#pragma unroll
  for (int mi = 0; mi < 2; ++mi)
#pragma unroll
    for (int ni = 0; ni < 4; ++ni) {
      int c = wc * 64 + ni * 16 + nloc;
      float bias = proj_b[c];
#pragma unroll
      for (int r = 0; r < 8; ++r) {
        int tok = mt * 128 + wr * 32 + mi * 16 + r + 8 * half;
        int id = tok >> 6, ih = (tok >> 3) & 7, iw = tok & 7;
        int df = ((wd * 8 + id) + 4) & 15;  // roll(+ss) after window reverse
        int hf = ((wh * 8 + ih) + 4) & 63;
        int wf = ((ww * 8 + iw) + 4) & 63;
        size_t idx = (((size_t)df * 64 + hf) * 64 + wf) * CDIM + c;
        out[idx] = x[idx] + acc.f[mi][ni][r] + bias;
      }
    }
}

// ----- K6: LN2 on d_out -> h2 bf16 ------------------------------------------
__global__ __launch_bounds__(256) void k_ln2(const float* __restrict__ xin,
                                             const float* __restrict__ g,
                                             const float* __restrict__ b,
                                             __bf16* __restrict__ h2) {
  int gw   = blockIdx.x * 8 + (threadIdx.x >> 5);
  int lane = threadIdx.x & 31;
  const float* src = xin + (size_t)gw * CDIM;
  float4 v = ((const float4*)src)[lane];
  float s  = v.x + v.y + v.z + v.w;
  float ss = v.x * v.x + v.y * v.y + v.z * v.z + v.w * v.w;
#pragma unroll
  for (int m = 16; m; m >>= 1) {
    s  += __shfl_xor(s, m, 32);
    ss += __shfl_xor(ss, m, 32);
  }
  float mean = s * (1.f / 128.f);
  float var  = ss * (1.f / 128.f) - mean * mean;
  float inv  = rsqrtf(var + 1e-5f);
  int c = lane * 4;
  union { __bf16 h4[4]; uint2 u; } pk;
  pk.h4[0] = (__bf16)((v.x - mean) * inv * g[c + 0] + b[c + 0]);
  pk.h4[1] = (__bf16)((v.y - mean) * inv * g[c + 1] + b[c + 1]);
  pk.h4[2] = (__bf16)((v.z - mean) * inv * g[c + 2] + b[c + 2]);
  pk.h4[3] = (__bf16)((v.w - mean) * inv * g[c + 3] + b[c + 3]);
  *(uint2*)(h2 + (size_t)gw * CDIM + c) = pk.u;
}

// ----- K7: MLP up-proj + exact GELU -> hid bf16 -----------------------------
__global__ __launch_bounds__(256) void k_mlp1(const __bf16* __restrict__ h2,
                                              const __bf16* __restrict__ W1,
                                              const float* __restrict__ b1,
                                              __bf16* __restrict__ hid) {
  int mt = blockIdx.x, nt = blockIdx.y;
  Acc acc;
  gemm128(h2 + (size_t)mt * 128 * CDIM, CDIM, W1, CDIM, nt * 128, CDIM, acc);

  int tid = threadIdx.x, wave = tid >> 5, wr = wave & 3, wc = wave >> 2;
  int lane = tid & 31, half = lane >> 4, nloc = lane & 15;
#pragma unroll
  for (int mi = 0; mi < 2; ++mi)
#pragma unroll
    for (int ni = 0; ni < 4; ++ni) {
      int n = nt * 128 + wc * 64 + ni * 16 + nloc;
      float bias = b1[n];
#pragma unroll
      for (int r = 0; r < 8; ++r) {
        int row = mt * 128 + wr * 32 + mi * 16 + r + 8 * half;
        float vv = acc.f[mi][ni][r] + bias;
        vv = 0.5f * vv * (1.f + erff(vv * 0.7071067811865476f));
        hid[(size_t)row * 512 + n] = (__bf16)vv;
      }
    }
}

// ----- K8: MLP down-proj + residual into d_out ------------------------------
__global__ __launch_bounds__(256) void k_mlp2(const __bf16* __restrict__ hid,
                                              const __bf16* __restrict__ W2,
                                              const float* __restrict__ b2,
                                              float* __restrict__ out) {
  int mt = blockIdx.x;
  Acc acc;
  gemm128(hid + (size_t)mt * 128 * 512, 512, W2, 512, 0, 512, acc);

  int tid = threadIdx.x, wave = tid >> 5, wr = wave & 3, wc = wave >> 2;
  int lane = tid & 31, half = lane >> 4, nloc = lane & 15;
#pragma unroll
  for (int mi = 0; mi < 2; ++mi)
#pragma unroll
    for (int ni = 0; ni < 4; ++ni) {
      int c = wc * 64 + ni * 16 + nloc;
      float bias = b2[c];
#pragma unroll
      for (int r = 0; r < 8; ++r) {
        int row = mt * 128 + wr * 32 + mi * 16 + r + 8 * half;
        size_t idx = (size_t)row * CDIM + c;
        out[idx] = out[idx] + acc.f[mi][ni][r] + bias;
      }
    }
}

// ---------------------------------------------------------------------------
extern "C" void kernel_launch(void* const* d_in, const int* in_sizes, int n_in,
                              void* d_out, int out_size, void* d_ws, size_t ws_size,
                              hipStream_t stream) {
  (void)in_sizes; (void)n_in; (void)out_size; (void)ws_size;
  const float* x         = (const float*)d_in[0];
  const float* attn_mask = (const float*)d_in[1];
  const int*   rel_index = (const int*)d_in[2];
  const float* rel_bias  = (const float*)d_in[3];
  const float* qkv_w     = (const float*)d_in[4];
  const float* qkv_b     = (const float*)d_in[5];
  const float* proj_w    = (const float*)d_in[6];
  const float* proj_b    = (const float*)d_in[7];
  const float* n1g       = (const float*)d_in[8];
  const float* n1b       = (const float*)d_in[9];
  const float* n2g       = (const float*)d_in[10];
  const float* n2b       = (const float*)d_in[11];
  const float* mlp_w1    = (const float*)d_in[12];
  const float* mlp_b1    = (const float*)d_in[13];
  const float* mlp_w2    = (const float*)d_in[14];
  const float* mlp_b2    = (const float*)d_in[15];
  float* out = (float*)d_out;

  // workspace carve-up (~85 MB), regions reused across sequential phases
  char* p = (char*)d_ws;
  auto take = [&](size_t bytes) {
    char* r = p;
    p += (bytes + 255) & ~(size_t)255;
    return r;
  };
  __bf16* Wq = (__bf16*)take((size_t)49152 * 2);   // [384][128] bf16
  __bf16* Wp = (__bf16*)take((size_t)16384 * 2);   // [128][128]
  __bf16* W1 = (__bf16*)take((size_t)65536 * 2);   // [512][128]
  __bf16* W2 = (__bf16*)take((size_t)65536 * 2);   // [128][512]
  float*  rpe = (float*)take((size_t)4 * 512 * 512 * 4);
  __bf16* xw  = (__bf16*)take((size_t)NWIN * NTOK * CDIM * 2);  // -> attn_out -> LN2-out
  __bf16* big = (__bf16*)take((size_t)TOKS * 512 * 2);          // q|k|vt -> MLP hidden
  __bf16* qb  = big;
  __bf16* kb  = big + (size_t)NWIN * NHEAD * NTOK * HD;
  __bf16* vt  = kb + (size_t)NWIN * NHEAD * NTOK * HD;  // transposed [hd][tok]
  __bf16* ao  = xw;   // attention output reuses xw region (xw dead after QKV)
  __bf16* h2  = xw;   // LN2 output reuses it again (ao dead after proj)
  __bf16* hid = big;  // MLP hidden reuses q/k/v region

  k_prep<<<(PREP_TOTAL + 255) / 256, 256, 0, stream>>>(
      qkv_w, proj_w, mlp_w1, mlp_w2, rel_index, rel_bias, Wq, Wp, W1, W2, rpe);
  k_ln1<<<TOKS / 8, 256, 0, stream>>>(x, n1g, n1b, xw);
  k_qkv<<<dim3(4, 3, NWIN), 256, 0, stream>>>(xw, Wq, qkv_b, qb, kb, vt);
  k_attn<<<NWIN * NHEAD, 256, 0, stream>>>(qb, kb, vt, rpe, attn_mask, ao);
  k_proj<<<dim3(4, 1, NWIN), 256, 0, stream>>>(ao, Wp, proj_b, x, out);
  k_ln2<<<TOKS / 8, 256, 0, stream>>>(out, n2g, n2b, h2);
  k_mlp1<<<dim3(TOKS / 128, 4), 256, 0, stream>>>(h2, W1, mlp_b1, hid);
  k_mlp2<<<dim3(TOKS / 128, 1), 256, 0, stream>>>(hid, W2, mlp_b2, out);
}
